// MaskedSelfAttention_52948356825790
// MI455X (gfx1250) — matrix operand order repaired and checked by reading, not compiled, not run on hardware
//
#include <hip/hip_runtime.h>

typedef __attribute__((ext_vector_type(16))) __bf16 v16bf;
typedef __attribute__((ext_vector_type(8)))  __bf16 v8bf;
typedef __attribute__((ext_vector_type(8)))  float  v8f;
typedef __attribute__((ext_vector_type(4)))  int    v4i;

#define DEVI static __device__ __forceinline__

constexpr int Bb = 4, Tt = 2048, Hh = 1024;
constexpr int BT = Bb * Tt;            // 8192
constexpr int SPITCH = 68;             // padded f32 pitch for S tile (bank spread)
constexpr int PPITCH = 72;             // padded bf16 pitch for P tile (16B-aligned)

DEVI __bf16 f2bf(float f) {
  unsigned u = __builtin_bit_cast(unsigned, f);
  unsigned r = u + 0x7FFFu + ((u >> 16) & 1u);   // round-to-nearest-even
  unsigned short h = (unsigned short)(r >> 16);
  return __builtin_bit_cast(__bf16, h);
}

union V16U { v16bf v; struct { v8bf lo, hi; } h; };

DEVI v8f wmma_bf16(v16bf a, v16bf b, v8f c) {
  return __builtin_amdgcn_wmma_f32_16x16x32_bf16(
      /*neg_a=*/false, a, /*neg_b=*/false, b,
      /*c_mod=*/(short)0, c, /*reuse_a=*/false, /*reuse_b=*/false);
}

// ---------------------------------------------------------------- convert f32 -> bf16
__global__ void cvt_f32_bf16(const float* __restrict__ s, __bf16* __restrict__ d, int n) {
  int i = (blockIdx.x * blockDim.x + threadIdx.x) * 4;
  if (i >= n) return;
  float4 v = *(const float4*)(s + i);
  unsigned short a = __builtin_bit_cast(unsigned short, f2bf(v.x));
  unsigned short b = __builtin_bit_cast(unsigned short, f2bf(v.y));
  unsigned short c = __builtin_bit_cast(unsigned short, f2bf(v.z));
  unsigned short e = __builtin_bit_cast(unsigned short, f2bf(v.w));
  int2 o;
  o.x = (int)((unsigned)a | ((unsigned)b << 16));
  o.y = (int)((unsigned)c | ((unsigned)e << 16));
  *(int2*)(d + i) = o;
}

// ---------------------------------------------------------------- QKV projection
// Each wave computes a 16(M=token) x 64(N=out-feature) tile of one of Q/K/V.
// B[k=h][n=o] = W[n][k] -> lane n reads W row n contiguously (no transpose).
__global__ __launch_bounds__(256) void qkv_proj(
    const __bf16* __restrict__ xb,
    const __bf16* __restrict__ wq, const float* __restrict__ bq,
    const __bf16* __restrict__ wk, const float* __restrict__ bk,
    const __bf16* __restrict__ wv, const float* __restrict__ bv,
    __bf16* __restrict__ Qb, __bf16* __restrict__ Kb, __bf16* __restrict__ VT) {
  const int lane  = threadIdx.x & 31;
  const int wid   = blockIdx.x * (blockDim.x >> 5) + (threadIdx.x >> 5);
  const int MT = BT / 16;      // 512
  const int NT = Hh / 64;      // 16
  const int mat   = wid / (MT * NT);
  if (mat > 2) return;
  const int rem   = wid % (MT * NT);
  const int mtile = rem / NT;
  const int nt64  = rem % NT;
  const __bf16* W    = (mat == 0) ? wq : (mat == 1) ? wk : wv;
  const float*  bias = (mat == 0) ? bq : (mat == 1) ? bk : bv;

  const int lhalf = (lane < 16) ? 0 : 1;
  const int l15   = lane & 15;

  const __bf16* arow = xb + (size_t)(mtile * 16 + l15) * Hh + lhalf * 8;

  v8f acc[4] = {};
  for (int k0 = 0; k0 < Hh; k0 += 32) {
    V16U a;
    a.h.lo = *(const v8bf*)(arow + k0);
    a.h.hi = *(const v8bf*)(arow + k0 + 16);
#pragma unroll
    for (int nt = 0; nt < 4; ++nt) {
      const __bf16* brow = W + (size_t)(nt64 * 64 + nt * 16 + l15) * Hh + lhalf * 8 + k0;
      V16U b;
      b.h.lo = *(const v8bf*)(brow);
      b.h.hi = *(const v8bf*)(brow + 16);
      acc[nt] = wmma_bf16(a.v, b.v, acc[nt]);
    }
  }

  if (mat < 2) {
    __bf16* outM = (mat == 0) ? Qb : Kb;
#pragma unroll
    for (int nt = 0; nt < 4; ++nt) {
      int n = nt64 * 64 + nt * 16 + l15;
      float bv_ = bias[n];
#pragma unroll
      for (int r = 0; r < 8; ++r) {
        int m = mtile * 16 + r + lhalf * 8;
        outM[(size_t)m * Hh + n] = f2bf(acc[nt][r] + bv_);
      }
    }
  } else {
    // store V transposed: VT[b][h][t], pitch T. C-frag rows are consecutive t.
    const int bidx = (mtile * 16) / Tt;
    const int t0   = (mtile * 16) % Tt + lhalf * 8;
#pragma unroll
    for (int nt = 0; nt < 4; ++nt) {
      int hcol = nt64 * 64 + nt * 16 + l15;
      float bv_ = bias[hcol];
      int d[4];
#pragma unroll
      for (int j = 0; j < 4; ++j) {
        unsigned short lo = __builtin_bit_cast(unsigned short, f2bf(acc[nt][2 * j]     + bv_));
        unsigned short hi = __builtin_bit_cast(unsigned short, f2bf(acc[nt][2 * j + 1] + bv_));
        d[j] = (int)((unsigned)lo | ((unsigned)hi << 16));
      }
      v4i dv = { d[0], d[1], d[2], d[3] };
      *(v4i*)(VT + ((size_t)(bidx * Hh + hcol) * Tt + t0)) = dv;
    }
  }
}

// ---------------------------------------------------------------- flash attention
// Block: 256 threads (8 waves), one batch b, 32 query rows. Loop 64-wide key blocks.
__global__ __launch_bounds__(256) void attn(
    const __bf16* __restrict__ Qb, const __bf16* __restrict__ Kb,
    const __bf16* __restrict__ VT, float* __restrict__ out) {
  __shared__ float  Ss[32][SPITCH];
  __shared__ __bf16 Ps[32][PPITCH];
  __shared__ float  mRow[32], lRow[32], fRow[32];

  const int tid   = threadIdx.x;
  const int lane  = tid & 31;
  const int w     = tid >> 5;               // wave 0..7
  const int lhalf = (lane < 16) ? 0 : 1;
  const int l15   = lane & 15;

  const int QT    = Tt / 32;                // 64
  const int bidx  = blockIdx.x / QT;
  // heaviest (largest qBase) tiles launch first: better tail behavior
  const int qBase = (QT - 1 - (blockIdx.x % QT)) * 32;
  const float NEGINF = -__builtin_inff();

  if (tid < 32) { mRow[tid] = NEGINF; lRow[tid] = 0.f; fRow[tid] = 1.f; }

  const int mi = w & 1;                     // S quadrant row-tile
  const int ni = w >> 1;                    // S quadrant col-tile (0..3)

  v8f acc[2][8] = {};                       // O slice: rows 0..31, cols [w*128, w*128+128)

  const __bf16* qrow = Qb + (size_t)(bidx * Tt + qBase + mi * 16 + l15) * Hh + lhalf * 8;
  __syncthreads();

  const int kbMax = (qBase + 31) >> 6;
  for (int kb = 0; kb <= kbMax; ++kb) {
    const int kStart = kb * 64;

    // --- S = Q K^T (each wave one 16x16 quadrant over full H) ---
    {
      v8f s = {};
      const __bf16* krow =
          Kb + (size_t)(bidx * Tt + kStart + ni * 16 + l15) * Hh + lhalf * 8;
      const bool pf = (kb < kbMax);
      for (int h0 = 0; h0 < Hh; h0 += 32) {
        V16U a, b;
        a.h.lo = *(const v8bf*)(qrow + h0);
        a.h.hi = *(const v8bf*)(qrow + h0 + 16);
        b.h.lo = *(const v8bf*)(krow + h0);
        b.h.hi = *(const v8bf*)(krow + h0 + 16);
        if (pf) __builtin_prefetch(krow + (size_t)64 * Hh + h0, 0, 3); // next K block
        s = wmma_bf16(a.v, b.v, s);
      }
      const float scale = 0.03125f;         // 1/sqrt(1024)
#pragma unroll
      for (int r = 0; r < 8; ++r)
        Ss[mi * 16 + r + lhalf * 8][ni * 16 + l15] = s[r] * scale;
    }
    __syncthreads();

    // --- causal mask + online softmax: 8 threads per row, all 256 threads ---
    {
      const int row = tid >> 3;             // 0..31
      const int c0  = (tid & 7) * 8;        // 8 cols per thread
      const int qg  = qBase + row;
      float mOld = mRow[row];

      float v[8];
      float4 va = *(const float4*)&Ss[row][c0];
      float4 vb = *(const float4*)&Ss[row][c0 + 4];
      v[0] = va.x; v[1] = va.y; v[2] = va.z; v[3] = va.w;
      v[4] = vb.x; v[5] = vb.y; v[6] = vb.z; v[7] = vb.w;

      float mLoc = NEGINF;
#pragma unroll
      for (int j = 0; j < 8; ++j) {
        if (kStart + c0 + j > qg) v[j] = NEGINF;
        mLoc = fmaxf(mLoc, v[j]);
      }
#pragma unroll
      for (int m = 1; m < 8; m <<= 1)
        mLoc = fmaxf(mLoc, __shfl_xor(mLoc, m, 32));
      float mNew = fmaxf(mOld, mLoc);

      float pe[8];
      float sLoc = 0.f;
#pragma unroll
      for (int j = 0; j < 8; ++j) {
        pe[j] = (v[j] == NEGINF) ? 0.f : __expf(v[j] - mNew);
        sLoc += pe[j];
      }
#pragma unroll
      for (int m = 1; m < 8; m <<= 1)
        sLoc += __shfl_xor(sLoc, m, 32);

      int d[4];
#pragma unroll
      for (int j = 0; j < 4; ++j) {
        unsigned short lo = __builtin_bit_cast(unsigned short, f2bf(pe[2 * j]));
        unsigned short hi = __builtin_bit_cast(unsigned short, f2bf(pe[2 * j + 1]));
        d[j] = (int)((unsigned)lo | ((unsigned)hi << 16));
      }
      v4i dv = { d[0], d[1], d[2], d[3] };
      *(v4i*)&Ps[row][c0] = dv;

      if ((tid & 7) == 0) {
        float fac = (mNew == NEGINF) ? 1.f : __expf(mOld - mNew);
        mRow[row] = mNew;
        lRow[row] = lRow[row] * fac + sLoc;
        fRow[row] = fac;
      }
    }
    __syncthreads();

    // --- rescale O accumulators, then O += P V  ---
#pragma unroll
    for (int a = 0; a < 2; ++a) {
      float f[8];
#pragma unroll
      for (int r = 0; r < 8; ++r) f[r] = fRow[a * 16 + r + lhalf * 8];
#pragma unroll
      for (int c = 0; c < 8; ++c)
#pragma unroll
        for (int r = 0; r < 8; ++r) acc[a][c][r] *= f[r];
    }

#pragma unroll
    for (int kk = 0; kk < 64; kk += 32) {
      V16U pf2[2];
#pragma unroll
      for (int a = 0; a < 2; ++a) {
        const __bf16* prow = &Ps[a * 16 + l15][kk + lhalf * 8];
        pf2[a].h.lo = *(const v8bf*)(prow);
        pf2[a].h.hi = *(const v8bf*)(prow + 16);
      }
#pragma unroll
      for (int c = 0; c < 8; ++c) {
        int hcol = w * 128 + c * 16 + l15;
        const __bf16* vrow =
            VT + (size_t)(bidx * Hh + hcol) * Tt + kStart + kk + lhalf * 8;
        V16U b;
        b.h.lo = *(const v8bf*)(vrow);
        b.h.hi = *(const v8bf*)(vrow + 16);
#pragma unroll
        for (int a = 0; a < 2; ++a) acc[a][c] = wmma_bf16(pf2[a].v, b.v, acc[a][c]);
      }
    }
    __syncthreads();
  }

  // --- normalize by row sums, write f32 output ---
#pragma unroll
  for (int a = 0; a < 2; ++a) {
    float inv[8];
#pragma unroll
    for (int r = 0; r < 8; ++r) {
      float l = lRow[a * 16 + r + lhalf * 8];
      inv[r] = (l > 0.f) ? (1.f / l) : 0.f;
    }
#pragma unroll
    for (int c = 0; c < 8; ++c) {
      int hcol = w * 128 + c * 16 + l15;
#pragma unroll
      for (int r = 0; r < 8; ++r) {
        int trow = qBase + a * 16 + r + lhalf * 8;
        out[(size_t)(bidx * Tt + trow) * Hh + hcol] = acc[a][c][r] * inv[r];
      }
    }
  }
}

// ---------------------------------------------------------------- launcher
extern "C" void kernel_launch(void* const* d_in, const int* in_sizes, int n_in,
                              void* d_out, int out_size, void* d_ws, size_t ws_size,
                              hipStream_t stream) {
  (void)in_sizes; (void)n_in; (void)out_size; (void)ws_size;
  const float* x  = (const float*)d_in[0];
  const float* Wq = (const float*)d_in[1];
  const float* bq = (const float*)d_in[2];
  const float* Wk = (const float*)d_in[3];
  const float* bk = (const float*)d_in[4];
  const float* Wv = (const float*)d_in[5];
  const float* bv = (const float*)d_in[6];
  float* out = (float*)d_out;

  const size_t XB = (size_t)BT * Hh * sizeof(__bf16);   // 16 MB
  const size_t WB = (size_t)Hh * Hh * sizeof(__bf16);   // 2 MB
  char* ws = (char*)d_ws;
  __bf16* xb  = (__bf16*)(ws);
  __bf16* wqb = (__bf16*)(ws + XB);
  __bf16* wkb = (__bf16*)(ws + XB + WB);
  __bf16* wvb = (__bf16*)(ws + XB + 2 * WB);
  __bf16* Qb  = (__bf16*)(ws + XB + 3 * WB);
  __bf16* Kb  = (__bf16*)(ws + 2 * XB + 3 * WB);
  __bf16* VT  = (__bf16*)(ws + 3 * XB + 3 * WB);

  const int nX = BT * Hh;        // 8,388,608
  const int nW = Hh * Hh;        // 1,048,576
  cvt_f32_bf16<<<nX / 4 / 256, 256, 0, stream>>>(x,  xb,  nX);
  cvt_f32_bf16<<<nW / 4 / 256, 256, 0, stream>>>(Wq, wqb, nW);
  cvt_f32_bf16<<<nW / 4 / 256, 256, 0, stream>>>(Wk, wkb, nW);
  cvt_f32_bf16<<<nW / 4 / 256, 256, 0, stream>>>(Wv, wvb, nW);

  // 3 matrices * 512 M-tiles * 16 N-tiles = 24576 waves / 8 waves-per-block
  qkv_proj<<<3072, 256, 0, stream>>>(xb, wqb, bq, wkb, bk, wvb, bv, Qb, Kb, VT);

  attn<<<Bb * (Tt / 32), 256, 0, stream>>>(Qb, Kb, VT, out);
}
